// VARS_D_18098992185936
// MI455X (gfx1250) — matrix operand order repaired and checked by reading, not compiled, not run on hardware
//
#include <hip/hip_runtime.h>

typedef __attribute__((ext_vector_type(16))) _Float16 v16h;
typedef __attribute__((ext_vector_type(8)))  _Float16 v8h;
typedef __attribute__((ext_vector_type(8)))  float    v8f;

#define DEV __device__ __forceinline__

#define Bb 32
#define Nn 1024
#define Cc_ 768
#define Hh 12
#define HD 64
#define Mm 128
#define LAM 0.1f
#define QSCALE 0.35355339059327373f   /* 64^-0.25 */
#define INV_SQRT_M 0.08838834764831845f /* 1/sqrt(128) */

DEV v8f wmma16(v16h a, v16h b, v8f c) {
  return __builtin_amdgcn_wmma_f32_16x16x32_f16(false, a, false, b, (short)0, c, false, false);
}

// ---- CDNA5 async global->LDS copies (ASYNCcnt-tracked, bypass VGPRs) ----
DEV void async_cp32(const _Float16* g, _Float16* l) {   // 32 bytes
  unsigned lo = (unsigned)(uintptr_t)l;                 // low 32 bits = LDS address
  unsigned long long ga = (unsigned long long)(uintptr_t)g;
  asm volatile("global_load_async_to_lds_b128 %0, %1, off\n\t"
               "global_load_async_to_lds_b128 %0, %1, off offset:16"
               :: "v"(lo), "v"(ga) : "memory");
}
DEV void async_cp16(const _Float16* g, _Float16* l) {   // 16 bytes
  asm volatile("global_load_async_to_lds_b128 %0, %1, off"
               :: "v"((unsigned)(uintptr_t)l), "v"((unsigned long long)(uintptr_t)g)
               : "memory");
}
DEV void async_wait0() { asm volatile("s_wait_asynccnt 0" ::: "memory"); }

// A fragment (16x32 f16): rows row0..row0+15, K cols k0..k0+31, from row-major lds (stride ld halves)
DEV v16h load_a_frag(const _Float16* p0, int row0, int k0, int ld) {
  int lane = threadIdx.x & 31;
  const _Float16* p = p0 + (size_t)(row0 + (lane & 15)) * ld + k0 + ((lane & 16) ? 16 : 0);
  v16h a;
#pragma unroll
  for (int i = 0; i < 8; ++i) { a[i] = p[i]; a[i + 8] = p[8 + i]; }
  return a;
}
// B fragment (32x16 f16): K rows k0..k0+31, cols col0..col0+15, from row-major [k][n] lds
DEV v16h load_b_frag(const _Float16* p0, int k0, int col0, int ld) {
  int lane = threadIdx.x & 31;
  const _Float16* p = p0 + (size_t)(k0 + ((lane & 16) ? 16 : 0)) * ld + col0 + (lane & 15);
  v16h b;
#pragma unroll
  for (int i = 0; i < 16; ++i) b[i] = p[(size_t)i * ld];
  return b;
}
// B fragment when staged transposed in LDS as [col][k] (contiguous k per column)
DEV v16h load_bT_frag(const _Float16* p0, int k0, int col0, int ld) {
  int lane = threadIdx.x & 31;
  const _Float16* p = p0 + (size_t)(col0 + (lane & 15)) * ld + k0 + ((lane & 16) ? 16 : 0);
  v16h b;
#pragma unroll
  for (int i = 0; i < 16; ++i) b[i] = p[i];
  return b;
}
// Store f32 C fragment to f16 LDS tile at (row0,col0)
DEV void store_tile_f16(_Float16* p0, int row0, int col0, int ld, v8f acc) {
  int lane = threadIdx.x & 31;
  int col = col0 + (lane & 15);
  int rb  = row0 + ((lane & 16) ? 8 : 0);
#pragma unroll
  for (int i = 0; i < 8; ++i) p0[(size_t)(rb + i) * ld + col] = (_Float16)acc[i];
}

DEV float softg(float z, float thr) {
  float u = fabsf(z) - thr;
  float g = 0.5f * u * (1.0f + erff(u * 0.7071067811865475f));
  float s = (z > 0.0f) ? 1.0f : ((z < 0.0f) ? -1.0f : 0.0f);
  return s * g;
}

// ---------------- conversions ----------------
__global__ void k_cvt(const float* __restrict__ s, _Float16* __restrict__ d, int n) {
  int i = blockIdx.x * 256 + threadIdx.x;
  if (i < n) d[i] = (_Float16)s[i];
}
__global__ void k_zero(float* __restrict__ d, int n) {
  int i = blockIdx.x * 256 + threadIdx.x;
  if (i < n) d[i] = 0.0f;
}

// ---------------- big weight GEMMs: out[M x Ncols] = A[M x K] * W[Ncols x K]^T ----------------
// Double-buffered async LDS staging. mode 0: qkv epilogue; mode 1: proj epilogue.
__global__ __launch_bounds__(256) void k_gemm_wgt(
    const _Float16* __restrict__ A, const _Float16* __restrict__ W, int K, int mode,
    _Float16* __restrict__ qh, _Float16* __restrict__ vh,
    const float* __restrict__ bias, float* __restrict__ out) {
  __shared__ __align__(32) _Float16 sA[2][128 * 32];
  __shared__ __align__(32) _Float16 sB[2][128 * 32]; // [col][k]
  const int tid = threadIdx.x, lane = tid & 31, w = tid >> 5;
  const int wr = w >> 1, wc = w & 1;
  const long arow0 = (long)blockIdx.x * 128;
  const long bcol0 = (long)blockIdx.y * 128;
  const int r2 = tid >> 1, seg = tid & 1;
  const _Float16* gA = A + (size_t)(arow0 + r2) * K + seg * 16;
  const _Float16* gW = W + (size_t)(bcol0 + r2) * K + seg * 16;
  _Float16* lA0 = &sA[0][r2 * 32 + seg * 16];
  _Float16* lA1 = &sA[1][r2 * 32 + seg * 16];
  _Float16* lB0 = &sB[0][r2 * 32 + seg * 16];
  _Float16* lB1 = &sB[1][r2 * 32 + seg * 16];
  v8f acc[2][4] = {};
  const int nsteps = K >> 5;
  async_cp32(gA, lA0);
  async_cp32(gW, lB0);
  for (int s = 0; s < nsteps; ++s) {
    const int cur = s & 1;
    async_wait0();
    __syncthreads();
    if (s + 1 < nsteps) {
      const size_t koff = (size_t)(s + 1) * 32;
      async_cp32(gA + koff, cur ? lA0 : lA1);
      async_cp32(gW + koff, cur ? lB0 : lB1);
    }
    const _Float16* cA = sA[cur];
    const _Float16* cB = sB[cur];
    v16h af0 = load_a_frag(cA, wr * 32, 0, 32);
    v16h af1 = load_a_frag(cA, wr * 32 + 16, 0, 32);
    v16h bf[4];
#pragma unroll
    for (int ct = 0; ct < 4; ++ct) bf[ct] = load_bT_frag(cB, 0, wc * 64 + ct * 16, 32);
#pragma unroll
    for (int ct = 0; ct < 4; ++ct) {
      acc[0][ct] = wmma16(af0, bf[ct], acc[0][ct]);
      acc[1][ct] = wmma16(af1, bf[ct], acc[1][ct]);
    }
  }
  const int nsub = lane & 15, mbase = (lane & 16) ? 8 : 0;
  if (mode == 0) {
#pragma unroll
    for (int rt = 0; rt < 2; ++rt)
#pragma unroll
      for (int ct = 0; ct < 4; ++ct)
#pragma unroll
        for (int i = 0; i < 8; ++i) {
          long R = arow0 + wr * 32 + rt * 16 + mbase + i;
          long Cg = bcol0 + wc * 64 + ct * 16 + nsub;
          float vv = acc[rt][ct][i];
          long b = R >> 10, n = R & 1023;
          if (Cg < 768) {
            long h = Cg >> 6, hd = Cg & 63;
            qh[((b * Hh + h) * Nn + n) * HD + hd] = (_Float16)(vv * QSCALE);
          } else {
            long c2 = Cg - 768, h = c2 >> 6, hd = c2 & 63;
            vh[((b * Hh + h) * Nn + n) * HD + hd] = (_Float16)vv;
          }
        }
  } else {
#pragma unroll
    for (int rt = 0; rt < 2; ++rt)
#pragma unroll
      for (int ct = 0; ct < 4; ++ct)
#pragma unroll
        for (int i = 0; i < 8; ++i) {
          long R = arow0 + wr * 32 + rt * 16 + mbase + i;
          long Cg = bcol0 + wc * 64 + ct * 16 + nsub;
          out[R * Cc_ + Cg] = acc[rt][ct][i] + bias[Cg];
        }
  }
}

// ---------------- qf = exp(q@rand - |q|^2/2)/sqrt(M), plus column sumsq ----------------
__global__ __launch_bounds__(256) void k_qf(
    const _Float16* __restrict__ qh, const _Float16* __restrict__ randh,
    _Float16* __restrict__ qfh, float* __restrict__ colnorm) {
  __shared__ __align__(32) _Float16 sQ[128 * 64];
  __shared__ __align__(32) _Float16 sR[64 * 128];
  __shared__ float snorm[128];
  __shared__ float cpart[128];
  const int tid = threadIdx.x, lane = tid & 31, w = tid >> 5;
  const int nt = blockIdx.x;
  const long bh = blockIdx.y;
  const int h = (int)(bh % Hh);
  const _Float16* qbase = qh + (bh * Nn + (long)nt * 128) * HD;
  const _Float16* rb = randh + (size_t)h * HD * Mm;
  for (int i = tid; i < 512; i += 256) async_cp32(qbase + (size_t)i * 16, sQ + (size_t)i * 16);
  for (int i = tid; i < 512; i += 256) async_cp32(rb + (size_t)i * 16, sR + (size_t)i * 16);
  if (tid < 128) cpart[tid] = 0.0f;
  async_wait0();
  __syncthreads();
  if (tid < 128) {
    float s = 0.0f;
#pragma unroll
    for (int d = 0; d < HD; ++d) { float q = (float)sQ[tid * HD + d]; s += q * q; }
    snorm[tid] = s;
  }
  __syncthreads();
  v8f acc[8] = {};
#pragma unroll
  for (int ks = 0; ks < 2; ++ks) {
    v16h a = load_a_frag(sQ, w * 16, ks * 32, HD);
#pragma unroll
    for (int ct = 0; ct < 8; ++ct) {
      v16h b = load_b_frag(sR, ks * 32, ct * 16, Mm);
      acc[ct] = wmma16(a, b, acc[ct]);
    }
  }
  const int nsub = lane & 15, mbase = (lane & 16) ? 8 : 0;
#pragma unroll
  for (int ct = 0; ct < 8; ++ct)
#pragma unroll
    for (int i = 0; i < 8; ++i) {
      int row = w * 16 + mbase + i;
      int col = ct * 16 + nsub;
      float val = expf(acc[ct][i] - 0.5f * snorm[row]) * INV_SQRT_M;
      qfh[(bh * Nn + (long)nt * 128 + row) * Mm + col] = (_Float16)val;
      atomicAdd(&cpart[col], val * val);
    }
  __syncthreads();
  if (tid < 128) atomicAdd(&colnorm[bh * Mm + tid], cpart[tid]);
}

__global__ void k_normqf(_Float16* __restrict__ qfh, const float* __restrict__ colnorm) {
  long i = (long)blockIdx.x * 256 + threadIdx.x;
  long bh = i >> 17;            // N*M = 131072
  int m = (int)(i & 127);
  float nv = fmaxf(sqrtf(colnorm[bh * Mm + m]), 1e-12f);
  qfh[i] = (_Float16)((float)qfh[i] / nv);
}

// ---------------- kk = qf^T qf  [128x128], inp = qf^T v [128x64] per (b,h) ----------------
// Double-buffered: qf/v rows go async into sB; qf^T staged via registers into sA.
__global__ __launch_bounds__(256) void k_kkinp(
    const _Float16* __restrict__ qfh, const _Float16* __restrict__ vh,
    float* __restrict__ kk, float* __restrict__ inp) {
  __shared__ __align__(32) _Float16 sA[2][128 * 32];  // qf^T: [m][t]
  __shared__ __align__(32) _Float16 sB[2][32 * 192];  // [t][ qf(128) | v(64) ]
  const int tid = threadIdx.x, lane = tid & 31, w = tid >> 5;
  const long bh = blockIdx.x;
  const _Float16* qf0 = qfh + bh * Nn * Mm;
  const _Float16* v0 = vh + bh * Nn * HD;
  const int tt = tid >> 3, seg = tid & 7;
  v8f acc[12] = {};
  // stage tile t0 into buffer buf
  auto stage = [&](int t0, int buf) {
    const _Float16* gq = qf0 + (size_t)(t0 + tt) * Mm + seg * 16;
    async_cp32(gq, &sB[buf][tt * 192 + seg * 16]);
    async_cp16(v0 + (size_t)(t0 + tt) * HD + seg * 8, &sB[buf][tt * 192 + 128 + seg * 8]);
    v16h q = *(const v16h*)gq;
#pragma unroll
    for (int j = 0; j < 16; ++j) sA[buf][(seg * 16 + j) * 32 + tt] = q[j];
  };
  stage(0, 0);
  for (int s = 0; s < 32; ++s) {
    const int cur = s & 1;
    async_wait0();
    __syncthreads();
    if (s + 1 < 32) stage((s + 1) * 32, cur ^ 1);
    v16h a = load_a_frag(sA[cur], w * 16, 0, 32);
#pragma unroll
    for (int ct = 0; ct < 12; ++ct) {
      v16h b = load_b_frag(sB[cur], 0, ct * 16, 192);
      acc[ct] = wmma16(a, b, acc[ct]);
    }
  }
  const int nsub = lane & 15, mbase = (lane & 16) ? 8 : 0;
#pragma unroll
  for (int ct = 0; ct < 12; ++ct)
#pragma unroll
    for (int i = 0; i < 8; ++i) {
      int m = w * 16 + mbase + i;
      int col = ct * 16 + nsub;
      float vv = acc[ct][i];
      if (col < 128) kk[(bh * Mm + m) * Mm + col] = vv;
      else inp[(bh * Mm + m) * HD + (col - 128)] = vv;
    }
}

// ---------------- ISTA: L, kk_l = K kk K, 8 soft-threshold iterations ----------------
__global__ __launch_bounds__(256) void k_ista(
    const float* __restrict__ kk, const float* __restrict__ inp,
    const float* __restrict__ lk, const float* __restrict__ lamp,
    const float* __restrict__ Lp, _Float16* __restrict__ xsh) {
  __shared__ __align__(32) unsigned char smem[65536];
  float* red = (float*)smem;                    // phase A only
  _Float16* kkb = (_Float16*)smem;              // [0,32K): kk f16, then tmp
  _Float16* klb = (_Float16*)(smem + 32768);    // [32K,64K): learned_k f16, then kk_l
  _Float16* xsb = (_Float16*)smem;              // [0,16K): xs f16 (after GEMMs)
  const int tid = threadIdx.x, lane = tid & 31, w = tid >> 5;
  const long bh = blockIdx.x;
  const float* kkg = kk + bh * Mm * Mm;
  const float* inpg = inp + bh * Mm * HD;
  // Phase A: L = max_m sum_p |kk[m][p]| + 1
  if (tid < 128) {
    float s = 0.0f;
    for (int j = 0; j < 128; ++j) s += fabsf(kkg[tid * 128 + j]);
    red[tid] = s;
  }
  __syncthreads();
  for (int st = 64; st > 0; st >>= 1) {
    if (tid < st) red[tid] = fmaxf(red[tid], red[tid + st]);
    __syncthreads();
  }
  float Lv = red[0] + 1.0f;
  float lamv = lamp[0] * LAM;
  float Ll = Lv / Lp[0];
  float invLl = 1.0f / Ll;
  float thr1 = lamv * invLl;
  __syncthreads();
  // Phase B: kk and learned_k to f16 LDS
  for (int i = tid; i < 16384; i += 256) { kkb[i] = (_Float16)kkg[i]; klb[i] = (_Float16)lk[i]; }
  __syncthreads();
  // GEMM1: tmp = kk @ learned_k
  {
    v8f acc[8] = {};
#pragma unroll
    for (int ks = 0; ks < 4; ++ks) {
      v16h a = load_a_frag(kkb, w * 16, ks * 32, 128);
#pragma unroll
      for (int ct = 0; ct < 8; ++ct) {
        v16h b = load_b_frag(klb, ks * 32, ct * 16, 128);
        acc[ct] = wmma16(a, b, acc[ct]);
      }
    }
    __syncthreads();
#pragma unroll
    for (int ct = 0; ct < 8; ++ct) store_tile_f16(kkb, w * 16, ct * 16, 128, acc[ct]);
    __syncthreads();
  }
  // GEMM2: kk_l = learned_k @ tmp  -> stored into klb
  {
    v8f acc[8] = {};
#pragma unroll
    for (int ks = 0; ks < 4; ++ks) {
      v16h a = load_a_frag(klb, w * 16, ks * 32, 128);
#pragma unroll
      for (int ct = 0; ct < 8; ++ct) {
        v16h b = load_b_frag(kkb, ks * 32, ct * 16, 128);
        acc[ct] = wmma16(a, b, acc[ct]);
      }
    }
    __syncthreads();
#pragma unroll
    for (int ct = 0; ct < 8; ++ct) store_tile_f16(klb, w * 16, ct * 16, 128, acc[ct]);
    __syncthreads();
  }
  // Park kk_l A-fragments (this wave's 16 rows, all K) in registers
  v16h akl[4];
#pragma unroll
  for (int ks = 0; ks < 4; ++ks) akl[ks] = load_a_frag(klb, w * 16, ks * 32, 128);
  // inp parked per-lane in C-fragment layout; xs initialized in LDS (reusing kkb region)
  const int nsub = lane & 15, mbase = (lane & 16) ? 8 : 0;
  float rin[4][8];
#pragma unroll
  for (int ct = 0; ct < 4; ++ct)
#pragma unroll
    for (int i = 0; i < 8; ++i) {
      int idx = (w * 16 + mbase + i) * HD + ct * 16 + nsub;
      float t = inpg[idx];
      rin[ct][i] = t;
      xsb[idx] = (_Float16)softg(t, lamv);
    }
  __syncthreads();
  // 8 ISTA iterations
  for (int it = 0; it < 8; ++it) {
    v8f a4[4] = {};
#pragma unroll
    for (int ks = 0; ks < 4; ++ks) {
#pragma unroll
      for (int ct = 0; ct < 4; ++ct) {
        v16h b = load_b_frag(xsb, ks * 32, ct * 16, HD);
        a4[ct] = wmma16(akl[ks], b, a4[ct]);
      }
    }
    __syncthreads();
#pragma unroll
    for (int ct = 0; ct < 4; ++ct)
#pragma unroll
      for (int i = 0; i < 8; ++i) {
        int idx = (w * 16 + mbase + i) * HD + ct * 16 + nsub;
        float z = (float)xsb[idx] - (a4[ct][i] - rin[ct][i]) * invLl;
        xsb[idx] = (_Float16)softg(z, thr1);
      }
    __syncthreads();
  }
  for (int i = tid; i < Mm * HD; i += 256) xsh[bh * Mm * HD + i] = xsb[i];
}

// ---------------- out = qf @ xs -> [B,N,C] interleaved f16 ----------------
__global__ __launch_bounds__(256) void k_outp(
    const _Float16* __restrict__ qfh, const _Float16* __restrict__ xsh,
    _Float16* __restrict__ oh) {
  __shared__ __align__(32) _Float16 sQ[128 * 128];
  __shared__ __align__(32) _Float16 sX[128 * 64];
  const int tid = threadIdx.x, lane = tid & 31, w = tid >> 5;
  const int nt = blockIdx.x;
  const long bh = blockIdx.y;
  const long b = bh / Hh;
  const int h = (int)(bh % Hh);
  const _Float16* q0 = qfh + (bh * Nn + (long)nt * 128) * Mm;
  const _Float16* x0 = xsh + bh * Mm * HD;
  for (int i = tid; i < 1024; i += 256) async_cp32(q0 + (size_t)i * 16, sQ + (size_t)i * 16);
  for (int i = tid; i < 512; i += 256) async_cp32(x0 + (size_t)i * 16, sX + (size_t)i * 16);
  async_wait0();
  __syncthreads();
  v8f acc[4] = {};
#pragma unroll
  for (int ks = 0; ks < 4; ++ks) {
    v16h a = load_a_frag(sQ, w * 16, ks * 32, Mm);
#pragma unroll
    for (int ct = 0; ct < 4; ++ct) {
      v16h bf = load_b_frag(sX, ks * 32, ct * 16, HD);
      acc[ct] = wmma16(a, bf, acc[ct]);
    }
  }
  const int nsub = lane & 15, mbase = (lane & 16) ? 8 : 0;
#pragma unroll
  for (int ct = 0; ct < 4; ++ct)
#pragma unroll
    for (int i = 0; i < 8; ++i) {
      long n = (long)nt * 128 + w * 16 + mbase + i;
      int d = ct * 16 + nsub;
      oh[(b * Nn + n) * Cc_ + (long)h * HD + d] = (_Float16)acc[ct][i];
    }
}

extern "C" void kernel_launch(void* const* d_in, const int* in_sizes, int n_in,
                              void* d_out, int out_size, void* d_ws, size_t ws_size,
                              hipStream_t stream) {
  (void)in_sizes; (void)n_in; (void)out_size; (void)ws_size;
  const float* x      = (const float*)d_in[0];
  const float* qkv_w  = (const float*)d_in[1];
  const float* proj_w = (const float*)d_in[2];
  const float* proj_b = (const float*)d_in[3];
  const float* rand_m = (const float*)d_in[4];
  const float* lk     = (const float*)d_in[5];
  const float* lam    = (const float*)d_in[6];
  const float* Lr     = (const float*)d_in[7];
  char* ws = (char*)d_ws;
  size_t off = 0;
  auto take = [&](size_t bytes) { char* p = ws + off; off += (bytes + 255) & ~(size_t)255; return p; };
  _Float16* xh   = (_Float16*)take(50331648);   // x f16 [32768,768]
  _Float16* wqh  = (_Float16*)take(2359296);    // qkv_w f16 [1536,768]
  _Float16* wph  = (_Float16*)take(1179648);    // proj_w f16 [768,768]
  _Float16* rh   = (_Float16*)take(196608);     // rand f16 [12,64,128]
  _Float16* qh   = (_Float16*)take(50331648);   // q f16 [B,H,N,64] (scaled)
  _Float16* vh   = (_Float16*)take(50331648);   // v f16 [B,H,N,64]
  _Float16* qfh  = (_Float16*)take(100663296);  // qf f16 [B,H,N,128]
  float*    cn   = (float*)take(196608);        // colnorm [B*H,128]
  float*    kkf  = (float*)take(25165824);      // kk f32 [B*H,128,128]
  float*    inpf = (float*)take(12582912);      // inp f32 [B*H,128,64]
  _Float16* xsh  = (_Float16*)take(6291456);    // xs f16 [B*H,128,64]
  _Float16* oh   = (_Float16*)take(50331648);   // out f16 [32768,768]
  float* out = (float*)d_out;

  k_cvt<<<(25165824 + 255) / 256, 256, 0, stream>>>(x, xh, 25165824);
  k_cvt<<<(1179648 + 255) / 256, 256, 0, stream>>>(qkv_w, wqh, 1179648);
  k_cvt<<<(589824 + 255) / 256, 256, 0, stream>>>(proj_w, wph, 589824);
  k_cvt<<<(98304 + 255) / 256, 256, 0, stream>>>(rand_m, rh, 98304);
  k_zero<<<192, 256, 0, stream>>>(cn, 49152);

  k_gemm_wgt<<<dim3(256, 12), 256, 0, stream>>>(xh, wqh, 768, 0, qh, vh, nullptr, nullptr);
  k_qf<<<dim3(8, 384), 256, 0, stream>>>(qh, rh, qfh, cn);
  k_normqf<<<196608, 256, 0, stream>>>(qfh, cn);
  k_kkinp<<<384, 256, 0, stream>>>(qfh, vh, kkf, inpf);
  k_ista<<<384, 256, 0, stream>>>(kkf, inpf, lk, lam, Lr, xsh);
  k_outp<<<dim3(8, 384), 256, 0, stream>>>(qfh, xsh, oh);
  k_gemm_wgt<<<dim3(256, 6), 256, 0, stream>>>(oh, wph, 768, 1, nullptr, nullptr, proj_b, out);
}
